// SpikeAdaptSC_DetV6d3_30099130810955
// MI455X (gfx1250) — matrix operand order repaired and checked by reading, not compiled, not run on hardware
//
#include <hip/hip_runtime.h>
#include <hip/hip_bf16.h>
#include <math.h>

// ---------------------------------------------------------------------------
// Types for CDNA5 WMMA / TDM
// ---------------------------------------------------------------------------
typedef __attribute__((ext_vector_type(16))) _Float16     v16h;
typedef __attribute__((ext_vector_type(8)))  float        v8f;
typedef __attribute__((ext_vector_type(4)))  unsigned int u32x4;
typedef __attribute__((ext_vector_type(8)))  int          i32x8;
typedef __attribute__((ext_vector_type(4)))  int          i32x4;

#define HWC   4096      // H*W (64*64)
#define WIMG  64
#define TSTEP 8
#define CS    64

__device__ __forceinline__ float actf(float v, int act) {
    switch (act) {
        case 1: return v > 0.f ? v : 0.2f * v;                 // leaky_relu(0.2)
        case 2: return v / (1.f + __expf(-v));                 // silu
        case 3: return 1.f / (1.f + __expf(-v));               // sigmoid
        case 4: return tanhf(v);                               // tanh
    }
    return v;
}

__device__ __forceinline__ unsigned hashu(unsigned x) {
    x ^= x >> 16; x *= 0x7feb352dU;
    x ^= x >> 15; x *= 0x846ca68bU;
    x ^= x >> 16; return x;
}
__device__ __forceinline__ float uni01(unsigned x) {
    return (float)(hashu(x) >> 8) * (1.0f / 16777216.0f);
}

// Pack one 16x16x32 f16 A-fragment from f32 LDS tile (row = M within tile).
__device__ __forceinline__ v16h packA32(const float (*As)[32], int mrow, int hi) {
    v16h a;
    #pragma unroll
    for (int v = 0; v < 8; ++v) {
        int kA = ((v < 4) ? 0 : 16) + ((v & 3) << 1) + (hi << 3);
        a[2 * v]     = (_Float16)As[mrow][kA];
        a[2 * v + 1] = (_Float16)As[mrow][kA + 1];
    }
    return a;
}
// Pack one 16x16x32 f16 B-fragment from LDS (col = N within tile).
__device__ __forceinline__ v16h packB(const _Float16 (*Bs)[130], int ncol, int hi) {
    v16h b;
    #pragma unroll
    for (int v = 0; v < 8; ++v) {
        int kB = ((v < 4) ? 0 : 16) + ((v & 3) << 1) + (hi << 3);
        b[2 * v]     = Bs[kB][ncol];
        b[2 * v + 1] = Bs[kB + 1][ncol];
    }
    return b;
}

// ---------------------------------------------------------------------------
// Generic implicit-GEMM conv (NCHW, 64x64 images) with fused bias+BN+act.
// Block: 256 threads (8 waves). Tile: M=64 x N=128 pixels, K staged 32/step.
// Weight tile (64 x 32 f32, row stride KTOT) is staged LDS-side by the
// Tensor Data Mover (tensor_load_to_lds, TENSORcnt); im2col tile is a manual
// gather+f16 convert. Each wave holds 2x2 accumulators -> 4 WMMAs / K-step.
// ---------------------------------------------------------------------------
__launch_bounds__(256)
__global__ void conv_wmma_kernel(const float* __restrict__ x,
                                 const float* __restrict__ w,
                                 const float* __restrict__ bias,
                                 const float* __restrict__ bnp,
                                 float* __restrict__ y,
                                 int CIN, int COUT, int KS, int PAD,
                                 int act, int useBN)
{
    __shared__ float    As32[64][32];   // [m][k]  64x32 f32, filled by TDM
    __shared__ _Float16 Bs[32][130];    // [k][n]  32x128 f16 (+pad)

    const int tid  = threadIdx.x;
    const int lane = tid & 31;
    const int wv   = tid >> 5;
    const int mtp  = wv & 1;                  // M half   (0..1) -> 32 rows
    const int nt   = wv >> 1;                 // N quarter(0..3) -> 32 pixels
    const int hi   = lane >> 4;               // lane half
    const int l16  = lane & 15;

    const int pix0  = blockIdx.x * 128;       // 128 pixels per block
    const int mbase = blockIdx.y * 64;        // 64 output channels per block
    const int KSQ   = KS * KS;
    const int KTOT  = CIN * KSQ;
    const int kstep = (KTOT + 31) >> 5;

    v8f acc00 = {0.f,0.f,0.f,0.f,0.f,0.f,0.f,0.f};
    v8f acc01 = acc00, acc10 = acc00, acc11 = acc00;

    const unsigned ldsA = (unsigned)(unsigned long long)(&As32[0][0]);

    for (int ks = 0; ks < kstep; ++ks) {
        const int k0 = ks << 5;

        // ---- stage A (weights) via Tensor Data Mover: 2-D tile 64(M)x32(K) --
        if (wv == 0) {
            unsigned long long ga =
                (unsigned long long)(w + (size_t)mbase * KTOT + k0);
            unsigned remK = (unsigned)(KTOT - k0);          // OOB K -> zero-fill
            unsigned remM = (unsigned)(COUT - mbase);       // OOB M -> zero-fill
            // D# group 0: count=1 | lds_addr | global_addr(57b) | type=2
            u32x4 g0 = { 1u,
                         ldsA,
                         (unsigned)(ga & 0xFFFFFFFFull),
                         (unsigned)((ga >> 32) & 0x1FFFFFFull) | (2u << 30) };
            // D# group 1: data_size=4B, tensor_dim0/1, tile_dim0=32 tile_dim1=64,
            //             tensor_dim0_stride = KTOT
            i32x8 g1 = { (int)(2u << 16),
                         (int)((remK & 0xFFFFu) << 16),
                         (int)((remK >> 16) | ((remM & 0xFFFFu) << 16)),
                         (int)((remM >> 16) | (32u << 16)),
                         (int)64u,
                         (int)KTOT,
                         0, 0 };
            i32x4 gz = { 0, 0, 0, 0 };
#if __clang_major__ >= 23
            i32x8 gz8 = { 0, 0, 0, 0, 0, 0, 0, 0 };
            __builtin_amdgcn_tensor_load_to_lds(g0, g1, gz, gz, gz8, 0);
#else
            __builtin_amdgcn_tensor_load_to_lds(g0, g1, gz, gz, 0);
#endif
            __builtin_amdgcn_s_wait_tensorcnt(0);
        }
        // ---- stage B (im2col) : 32x128 f16, 16 elems/thread, pixel-contig ----
        #pragma unroll
        for (int i = 0; i < 16; ++i) {
            int e  = i * 256 + tid;
            int kk = e >> 7;
            int j  = e & 127;
            int k  = k0 + kk;
            float v = 0.f;
            if (k < KTOT) {
                int cin = k / KSQ;
                int r   = k - cin * KSQ;
                int kh  = r / KS;
                int kw  = r - kh * KS;
                int p   = pix0 + j;
                int b   = p >> 12;
                int rem = p & 4095;
                int oh  = rem >> 6;
                int ow  = rem & 63;
                int ih  = oh + kh - PAD;
                int iw  = ow + kw - PAD;
                if (ih >= 0 && ih < 64 && iw >= 0 && iw < 64)
                    v = x[(((size_t)(b * CIN + cin)) << 12) + (ih << 6) + iw];
            }
            Bs[kk][j] = (_Float16)v;
        }
        __syncthreads();

        // ---- 2 A frags x 2 B frags -> 4 WMMAs (register tile reuse) ----
        v16h a0 = packA32(As32, mtp * 32 + l16,      hi);
        v16h a1 = packA32(As32, mtp * 32 + 16 + l16, hi);
        v16h b0 = packB(Bs, nt * 32 + l16,       hi);
        v16h b1 = packB(Bs, nt * 32 + 16 + l16,  hi);

        acc00 = __builtin_amdgcn_wmma_f32_16x16x32_f16(false, a0, false, b0, (short)0, acc00, false, false);
        acc01 = __builtin_amdgcn_wmma_f32_16x16x32_f16(false, a0, false, b1, (short)0, acc01, false, false);
        acc10 = __builtin_amdgcn_wmma_f32_16x16x32_f16(false, a1, false, b0, (short)0, acc10, false, false);
        acc11 = __builtin_amdgcn_wmma_f32_16x16x32_f16(false, a1, false, b1, (short)0, acc11, false, false);
        __syncthreads();
    }

    // ---- epilogue: bias + BN + activation, one store per output element ----
    #pragma unroll
    for (int ti = 0; ti < 2; ++ti) {
        #pragma unroll
        for (int tj = 0; tj < 2; ++tj) {
            const v8f& acc = (ti == 0) ? (tj == 0 ? acc00 : acc01)
                                       : (tj == 0 ? acc10 : acc11);
            const int p   = pix0 + nt * 32 + tj * 16 + l16;
            const int b   = p >> 12;
            const int rem = p & 4095;
            #pragma unroll
            for (int r = 0; r < 8; ++r) {
                int mg = mbase + mtp * 32 + ti * 16 + (hi << 3) + r; // C/D: M=r+8*hi
                if (mg < COUT) {
                    float v = acc[r] + bias[mg];
                    if (useBN) {
                        float g  = bnp[mg];
                        float bt = bnp[COUT + mg];
                        float mu = bnp[2 * COUT + mg];
                        float vr = bnp[3 * COUT + mg];
                        v = g * rsqrtf(vr + 1e-5f) * (v - mu) + bt;
                    }
                    v = actf(v, act);
                    y[(((size_t)(b * COUT + mg)) << 12) + rem] = v;
                }
            }
        }
    }
}

// ---------------------------------------------------------------------------
// I-LIF recurrence with per-timestep MPBN.  One thread per (b,c,h,w).
// ---------------------------------------------------------------------------
__global__ void lif_kernel(const float* __restrict__ xp,
                           const float* __restrict__ mpbn,      // [T,4,64]
                           const float* __restrict__ thr,       // [64]
                           const float* __restrict__ betaRaw,   // [64]
                           const float* __restrict__ runMean,   // [1]
                           float* __restrict__ spikes,          // [T, B*64*HW]
                           float* __restrict__ raw,             // [T, B*64*HW]
                           int total)
{
    int i = blockIdx.x * blockDim.x + threadIdx.x;
    if (i >= total) return;
    int c = (i >> 12) & 63;
    float xv   = xp[i];
    float beta = 1.f / (1.f + __expf(-betaRaw[c]));
    float thb  = fabsf(thr[c]) * fmaxf(runMean[0], 0.1f);
    float thd  = thb + 1e-6f;
    float mem  = 0.f;
    for (int t = 0; t < TSTEP; ++t) {
        const float* mp = mpbn + t * 256;
        float g  = mp[c], bt = mp[64 + c], mu = mp[128 + c], vr = mp[192 + c];
        float xt = g * rsqrtf(vr + 1e-5f) * (xv - mu) + bt;
        mem = beta * mem + xt;
        float sp = rintf(fminf(fmaxf(mem / thd, 0.f), 4.f));
        spikes[(size_t)t * total + i] = sp;
        raw[(size_t)t * total + i]   = mem;
        mem -= sp * thb;
    }
}

// mean spike rate over T, plus constant noise channel 64 -> [B,65,HW]
__global__ void meanrate_kernel(const float* __restrict__ spikes,
                                float* __restrict__ out, int n)
{
    int i = blockIdx.x * blockDim.x + threadIdx.x;
    if (i >= n) return;
    int hw = i & 4095;
    int c  = (i >> 12) % 65;
    int b  = i / (65 * 4096);
    if (c == 64) { out[i] = 0.05f; return; }
    size_t base = (((size_t)(b * CS + c)) << 12) + hw;
    float s = 0.f;
    for (int t = 0; t < TSTEP; ++t) s += spikes[(size_t)t * 2097152 + base];
    out[i] = s * 0.125f;
}

// scorer head: 1x1 conv 32->1 + sigmoid
__global__ void scorer2_kernel(const float* __restrict__ h1,
                               const float* __restrict__ w2,
                               const float* __restrict__ b2,
                               float* __restrict__ imp, int n)
{
    int i = blockIdx.x * blockDim.x + threadIdx.x;
    if (i >= n) return;
    int b  = i >> 12;
    int hw = i & 4095;
    float s = b2[0];
    for (int c = 0; c < 32; ++c)
        s += h1[(((size_t)(b * 32 + c)) << 12) + hw] * w2[c];
    imp[i] = 1.f / (1.f + __expf(-s));
}

// deterministic top-k mask via threshold bisection (k = 3072 of 4096)
__global__ void topk_kernel(const float* __restrict__ imp, float* __restrict__ mask)
{
    __shared__ int cnt;
    const int b = blockIdx.x;
    const float* v = imp + ((size_t)b << 12);
    float lo = 0.f, hi = 1.f;
    for (int it = 0; it < 24; ++it) {
        float mid = 0.5f * (lo + hi);
        if (threadIdx.x == 0) cnt = 0;
        __syncthreads();
        int c = 0;
        for (int j = threadIdx.x; j < 4096; j += 256) c += (v[j] > mid);
        atomicAdd(&cnt, c);
        __syncthreads();
        if (cnt >= 3072) lo = mid; else hi = mid;
        __syncthreads();
    }
    for (int j = threadIdx.x; j < 4096; j += 256)
        mask[((size_t)b << 12) + j] = (v[j] > lo) ? 1.f : 0.f;
}

// masked spikes -> BSC channel noise -> softmax-weighted temporal avg (/4)
__global__ void spike_avg_kernel(const float* __restrict__ spikes,
                                 const float* __restrict__ mask,
                                 const float* __restrict__ tw,
                                 float* __restrict__ out, int n)
{
    int i = blockIdx.x * blockDim.x + threadIdx.x;
    if (i >= n) return;
    int hw = i & 4095;
    int b  = i >> 18;                   // / (64*4096)
    float m = mask[((size_t)b << 12) + hw];
    float e[TSTEP], s = 0.f;
    for (int t = 0; t < TSTEP; ++t) { e[t] = __expf(tw[t]); s += e[t]; }
    float inv = 1.f / s, acc = 0.f;
    for (int t = 0; t < TSTEP; ++t) {
        float sp = spikes[(size_t)t * n + i] * m;
        unsigned id = (unsigned)(t * n + i);
        float u1 = uni01(id * 2u + 0x1234567u);
        float u2 = uni01(id * 2u + 0x89abcd1u);
        float flip = (u1 < 0.05f) ? 1.f : 0.f;
        float pert = (u2 > 0.5f) ? 1.f : -1.f;
        float recv = fminf(fmaxf(sp + flip * pert, 0.f), 4.f);
        acc += e[t] * inv * recv;
    }
    out[i] = acc * 0.25f;   // / MAX_LEVEL
}

// masked mems -> gaussian noise -> softmax-weighted temporal avg
__global__ void mem_avg_kernel(const float* __restrict__ mems,
                               const float* __restrict__ mask,
                               const float* __restrict__ tw,
                               float* __restrict__ out, int n)
{
    int i = blockIdx.x * blockDim.x + threadIdx.x;
    if (i >= n) return;
    int hw = i & 4095;
    int b  = i >> 17;                   // / (32*4096)
    float m = mask[((size_t)b << 12) + hw];
    float e[TSTEP], s = 0.f;
    for (int t = 0; t < TSTEP; ++t) { e[t] = __expf(tw[t]); s += e[t]; }
    float inv = 1.f / s, acc = 0.f;
    for (int t = 0; t < TSTEP; ++t) {
        float mv = mems[(size_t)t * n + i] * m;
        unsigned id = (unsigned)(t * n + i);
        float u1 = fmaxf(uni01(id * 2u + 0xdeadbefu), 1e-7f);
        float u2 = uni01(id * 2u + 0xc0ffee1u);
        float g  = sqrtf(-2.f * __logf(u1)) * __cosf(6.2831853f * u2);
        acc += e[t] * inv * (mv + g * 0.025f);
    }
    out[i] = acc;
}

// concat [B,256,HW] + [B,256,HW] -> [B,512,HW]
__global__ void concat_kernel(const float* __restrict__ a,
                              const float* __restrict__ b,
                              float* __restrict__ out, int n)
{
    int i = blockIdx.x * blockDim.x + threadIdx.x;
    if (i >= n) return;
    int hw = i & 4095;
    int c  = (i >> 12) & 511;
    int bb = i >> 21;
    out[i] = (c < 256)
        ? a[(((size_t)(bb * 256 + c)) << 12) + hw]
        : b[(((size_t)(bb * 256 + (c - 256))) << 12) + hw];
}

// fused = gate*sf + (1-gate)*mf
__global__ void fuse_kernel(const float* __restrict__ g,
                            const float* __restrict__ sf,
                            const float* __restrict__ mf,
                            float* __restrict__ out, int n)
{
    int i = blockIdx.x * blockDim.x + threadIdx.x;
    if (i >= n) return;
    float gg = g[i];
    out[i] = gg * sf[i] + (1.f - gg) * mf[i];
}

// ---------------------------------------------------------------------------
extern "C" void kernel_launch(void* const* d_in, const int* in_sizes, int n_in,
                              void* d_out, int out_size, void* d_ws, size_t ws_size,
                              hipStream_t stream)
{
    const float* feat      = (const float*)d_in[0];
    const float* proj_w    = (const float*)d_in[1];
    const float* proj_b    = (const float*)d_in[2];
    const float* proj_bn   = (const float*)d_in[3];
    const float* mpbn      = (const float*)d_in[4];
    const float* lif_thr   = (const float*)d_in[5];
    const float* lif_beta  = (const float*)d_in[6];
    const float* lif_rm    = (const float*)d_in[7];
    const float* memenc_w  = (const float*)d_in[8];
    const float* memenc_b  = (const float*)d_in[9];
    const float* memenc_bn = (const float*)d_in[10];
    const float* sc_w1     = (const float*)d_in[11];
    const float* sc_b1     = (const float*)d_in[12];
    const float* sc_bn     = (const float*)d_in[13];
    const float* sc_w2     = (const float*)d_in[14];
    const float* sc_b2     = (const float*)d_in[15];
    const float* spike_tw  = (const float*)d_in[16];
    const float* mem_tw    = (const float*)d_in[17];
    const float* sdec_w    = (const float*)d_in[18];
    const float* sdec_b    = (const float*)d_in[19];
    const float* sdec_bn   = (const float*)d_in[20];
    const float* mdec_w    = (const float*)d_in[21];
    const float* mdec_b    = (const float*)d_in[22];
    const float* mdec_bn   = (const float*)d_in[23];
    const float* gate_w    = (const float*)d_in[24];
    const float* gate_b    = (const float*)d_in[25];
    const float* blend_w   = (const float*)d_in[26];
    const float* blend_b   = (const float*)d_in[27];
    const float* blend_bn  = (const float*)d_in[28];
    float* out = (float*)d_out;

    // ---- workspace layout (floats) ----
    float* ws = (float*)d_ws;
    size_t o = 0;
    float* xp     = ws + o; o += 2097152;   // [8,64,HW]
    float* spikes = ws + o; o += 16777216;  // [T,8,64,HW]
    float* raw    = ws + o; o += 16777216;  // [T,8,64,HW]
    float* mems   = ws + o; o += 8388608;   // [T,8,32,HW]
    float* scin   = ws + o; o += 2129920;   // [8,65,HW]
    float* h1     = ws + o; o += 1048576;   // [8,32,HW]
    float* imp    = ws + o; o += 32768;     // [8,HW]
    float* msk    = ws + o; o += 32768;     // [8,HW]
    float* spavg  = ws + o; o += 2097152;   // [8,64,HW]
    float* memavg = ws + o; o += 1048576;   // [8,32,HW]
    float* sfeat  = ws + o; o += 8388608;   // [8,256,HW]
    float* mfeat  = ws + o; o += 8388608;   // [8,256,HW]
    float* cat512 = raw;                    // reuse (raw dead after memenc)
    float* gateb  = spikes;                 // reuse (spikes dead after spike_avg)
    float* fused  = spikes + 8388608;

    // 1) proj conv 3x3 256->64 + BN
    conv_wmma_kernel<<<dim3(8 * HWC / 128, 1), 256, 0, stream>>>(
        feat, proj_w, proj_b, proj_bn, xp, 256, 64, 3, 1, 0, 1);

    // 2) MPBN + I-LIF recurrence
    lif_kernel<<<2097152 / 256, 256, 0, stream>>>(
        xp, mpbn, lif_thr, lif_beta, lif_rm, spikes, raw, 2097152);

    // 3) memenc 1x1 64->32 + BN + tanh over T*B=64 images
    conv_wmma_kernel<<<dim3(64 * HWC / 128, 1), 256, 0, stream>>>(
        raw, memenc_w, memenc_b, memenc_bn, mems, 64, 32, 1, 0, 4, 1);

    // 4) mean rate + noise channel -> [8,65,HW]
    meanrate_kernel<<<(2129920 + 255) / 256, 256, 0, stream>>>(spikes, scin, 2129920);

    // 5) scorer conv1 3x3 65->32 + BN + leaky
    conv_wmma_kernel<<<dim3(8 * HWC / 128, 1), 256, 0, stream>>>(
        scin, sc_w1, sc_b1, sc_bn, h1, 65, 32, 3, 1, 1, 1);

    // 6) scorer head 1x1 32->1 + sigmoid
    scorer2_kernel<<<32768 / 256, 256, 0, stream>>>(h1, sc_w2, sc_b2, imp, 32768);

    // 7) top-k (k=3072) mask per image
    topk_kernel<<<8, 256, 0, stream>>>(imp, msk);

    // 8) masked spikes + BSC noise -> weighted temporal avg / 4
    spike_avg_kernel<<<2097152 / 256, 256, 0, stream>>>(spikes, msk, spike_tw, spavg, 2097152);

    // 9) masked mems + gaussian noise -> weighted temporal avg
    mem_avg_kernel<<<1048576 / 256, 256, 0, stream>>>(mems, msk, mem_tw, memavg, 1048576);

    // 10) sdec conv 3x3 64->256 + BN + silu
    conv_wmma_kernel<<<dim3(8 * HWC / 128, 4), 256, 0, stream>>>(
        spavg, sdec_w, sdec_b, sdec_bn, sfeat, 64, 256, 3, 1, 2, 1);

    // 11) mdec conv 3x3 32->256 + BN + silu
    conv_wmma_kernel<<<dim3(8 * HWC / 128, 4), 256, 0, stream>>>(
        memavg, mdec_w, mdec_b, mdec_bn, mfeat, 32, 256, 3, 1, 2, 1);

    // 12) concat -> [8,512,HW]
    concat_kernel<<<16777216 / 256, 256, 0, stream>>>(sfeat, mfeat, cat512, 16777216);

    // 13) gate conv 1x1 512->256 + sigmoid (no BN)
    conv_wmma_kernel<<<dim3(8 * HWC / 128, 4), 256, 0, stream>>>(
        cat512, gate_w, gate_b, gate_b, gateb, 512, 256, 1, 0, 3, 0);

    // 14) gated fusion
    fuse_kernel<<<8388608 / 256, 256, 0, stream>>>(gateb, sfeat, mfeat, fused, 8388608);

    // 15) blend conv 3x3 256->256 + BN + silu -> output
    conv_wmma_kernel<<<dim3(8 * HWC / 128, 4), 256, 0, stream>>>(
        fused, blend_w, blend_b, blend_bn, out, 256, 256, 3, 1, 2, 1);
}
